// RecurrentRunner_8564164788390
// MI455X (gfx1250) — compile-verified, matrix operand order checked by reading, not run on hardware
//
#include <hip/hip_runtime.h>
#include <math.h>

typedef __bf16 bf16_t;
typedef __attribute__((ext_vector_type(16))) __bf16 v16bf;
typedef __attribute__((ext_vector_type(8)))  __bf16 v8bf;
typedef __attribute__((ext_vector_type(8)))  float  v8f;

#define D_IN  128
#define H_DIM 256

__device__ __forceinline__ float fast_tanhf(float x) {
#if __has_builtin(__builtin_amdgcn_tanhf)
    return __builtin_amdgcn_tanhf(x);
#else
    return tanhf(x);
#endif
}

// ---------------------------------------------------------------------------
// Phase 0: split W1[:, :128] into bf16 hi/lo planes, stored transposed-friendly
// as [n][k] (row n of W1 = column n of B), so the GEMM's B-fragment loads are
// contiguous 16B chunks.
// ---------------------------------------------------------------------------
__global__ __launch_bounds__(256) void prep_w1_kernel(
    const float* __restrict__ W1, bf16_t* __restrict__ Bhi, bf16_t* __restrict__ Blo)
{
    int idx = blockIdx.x * blockDim.x + threadIdx.x;   // 0 .. 256*128-1
    int n = idx >> 7;
    int k = idx & 127;
    float x = W1[(size_t)n * (D_IN + 1) + k];
    bf16_t h = (bf16_t)x;
    Bhi[idx] = h;
    Blo[idx] = (bf16_t)(x - (float)h);
}

// ---------------------------------------------------------------------------
// Phase 1: U = X @ W1[:, :128]^T + b1   ([T x 128] x [128 x 256] -> [T x 256])
// One wave per 16-row stripe; 16 accumulator tiles cover all 256 columns so X
// is streamed exactly once. bf16 hi/lo split: 3 WMMAs per (ktile, ntile).
// ---------------------------------------------------------------------------
__global__ __launch_bounds__(32) void gemm_u_kernel(
    const float*  __restrict__ X,
    const bf16_t* __restrict__ Bhi,
    const bf16_t* __restrict__ Blo,
    const float*  __restrict__ b1,
    float*        __restrict__ U)
{
    const int lane = threadIdx.x;            // 0..31, wave32
    const int m0   = blockIdx.x * 16;        // row tile base
    const int mrow = m0 + (lane & 15);       // A-frag: lane -> row
    const int koff = (lane >> 4) * 8;        // A/B-frag K offset per half-wave
    const int ncol = lane & 15;              // B/C-frag: lane -> column within tile

    v8f acc[16];
    #pragma unroll
    for (int nt = 0; nt < 16; ++nt) {
        float bias = b1[nt * 16 + ncol];
        acc[nt] = (v8f){bias, bias, bias, bias, bias, bias, bias, bias};
    }

    const float* xrow = X + (size_t)mrow * D_IN;

    #pragma unroll
    for (int kt = 0; kt < 4; ++kt) {
        const int kb = kt * 32 + koff;
        // A fragment: elems 0..7 <- K kb..kb+7, elems 8..15 <- K kb+16..kb+23
        float4 x0 = *(const float4*)(xrow + kb);
        float4 x1 = *(const float4*)(xrow + kb + 4);
        float4 x2 = *(const float4*)(xrow + kb + 16);
        float4 x3 = *(const float4*)(xrow + kb + 20);
        float xf[16] = {x0.x, x0.y, x0.z, x0.w,  x1.x, x1.y, x1.z, x1.w,
                        x2.x, x2.y, x2.z, x2.w,  x3.x, x3.y, x3.z, x3.w};
        v16bf ahi, alo;
        #pragma unroll
        for (int e = 0; e < 16; ++e) {
            bf16_t h = (bf16_t)xf[e];
            ahi[e] = h;
            alo[e] = (bf16_t)(xf[e] - (float)h);
        }

        #pragma unroll
        for (int nt = 0; nt < 16; ++nt) {
            const int n = nt * 16 + ncol;
            const bf16_t* bh = Bhi + (size_t)n * D_IN + kb;
            const bf16_t* bl = Blo + (size_t)n * D_IN + kb;
            v8bf bh0 = *(const v8bf*)(bh);
            v8bf bh1 = *(const v8bf*)(bh + 16);
            v8bf bl0 = *(const v8bf*)(bl);
            v8bf bl1 = *(const v8bf*)(bl + 16);
            v16bf bhi, blo;
            #pragma unroll
            for (int e = 0; e < 8; ++e) {
                bhi[e] = bh0[e]; bhi[e + 8] = bh1[e];
                blo[e] = bl0[e]; blo[e + 8] = bl1[e];
            }
            acc[nt] = __builtin_amdgcn_wmma_f32_16x16x32_bf16(
                false, ahi, false, bhi, (short)0, acc[nt], false, false);
            acc[nt] = __builtin_amdgcn_wmma_f32_16x16x32_bf16(
                false, ahi, false, blo, (short)0, acc[nt], false, false);
            acc[nt] = __builtin_amdgcn_wmma_f32_16x16x32_bf16(
                false, alo, false, bhi, (short)0, acc[nt], false, false);
        }
    }

    // C/D layout: VGPR j -> M = j (lanes 0-15) / j+8 (lanes 16-31), lane -> N
    const int rbase = m0 + ((lane >> 4) << 3);
    #pragma unroll
    for (int j = 0; j < 8; ++j) {
        const size_t row = (size_t)(rbase + j);
        #pragma unroll
        for (int nt = 0; nt < 16; ++nt)
            U[row * H_DIM + nt * 16 + ncol] = acc[nt][j];
    }
}

// ---------------------------------------------------------------------------
// Phase 2: the irreducible sequential scan. One wave32; 8 hidden units/lane.
//   z_{t+1} = tanh( sum_n W2[n]*tanh(U[t][n] + c[n]*z_t) + b2 )
// Next step's 1KB U row is prefetched into registers (2 x b128 per lane)
// before the current step's dependent math.
// ---------------------------------------------------------------------------
__global__ __launch_bounds__(32) void scan_kernel(
    const float* __restrict__ U,
    const float* __restrict__ W1,
    const float* __restrict__ W2,
    const float* __restrict__ b2,
    float*       __restrict__ out,
    int T)
{
    const int lane = threadIdx.x;
    float c[8], w2v[8];
    #pragma unroll
    for (int i = 0; i < 8; ++i) {
        int n = lane * 8 + i;
        c[i]   = W1[(size_t)n * (D_IN + 1) + D_IN];   // z-coefficient column
        w2v[i] = W2[n];
    }
    const float bias2 = b2[0];
    if (lane == 0) out[0] = 0.0f;

    const float4* u0 = (const float4*)U + lane * 2;   // 64 float4 per row
    float4 ua = u0[0], ub = u0[1];
    float z = 0.0f;

    for (int t = 0; t < T - 1; ++t) {
        // prefetch row t+1 (row T-1 exists: U has T rows)
        const float4* un = (const float4*)(U + (size_t)(t + 1) * H_DIM) + lane * 2;
        float4 na = un[0], nb = un[1];

        float v;
        v =          w2v[0] * fast_tanhf(fmaf(c[0], z, ua.x));
        v = fmaf(w2v[1], fast_tanhf(fmaf(c[1], z, ua.y)), v);
        v = fmaf(w2v[2], fast_tanhf(fmaf(c[2], z, ua.z)), v);
        v = fmaf(w2v[3], fast_tanhf(fmaf(c[3], z, ua.w)), v);
        v = fmaf(w2v[4], fast_tanhf(fmaf(c[4], z, ub.x)), v);
        v = fmaf(w2v[5], fast_tanhf(fmaf(c[5], z, ub.y)), v);
        v = fmaf(w2v[6], fast_tanhf(fmaf(c[6], z, ub.z)), v);
        v = fmaf(w2v[7], fast_tanhf(fmaf(c[7], z, ub.w)), v);

        // wave32 butterfly reduction
        #pragma unroll
        for (int off = 16; off > 0; off >>= 1)
            v += __shfl_xor(v, off, 32);

        z = fast_tanhf(v + bias2);
        if (lane == 0) out[t + 1] = z;

        ua = na; ub = nb;
    }
}

// ---------------------------------------------------------------------------
extern "C" void kernel_launch(void* const* d_in, const int* in_sizes, int n_in,
                              void* d_out, int out_size, void* d_ws, size_t ws_size,
                              hipStream_t stream)
{
    const float* X  = (const float*)d_in[0];
    const float* W1 = (const float*)d_in[1];
    const float* b1 = (const float*)d_in[2];
    const float* W2 = (const float*)d_in[3];
    const float* b2 = (const float*)d_in[4];
    float* out = (float*)d_out;

    const int T = in_sizes[0] / D_IN;      // 262144

    // workspace layout: U [T*256 f32] | Bhi [256*128 bf16] | Blo [256*128 bf16]
    char*   ws  = (char*)d_ws;
    float*  U   = (float*)ws;
    size_t  ub  = (size_t)T * H_DIM * sizeof(float);
    bf16_t* Bhi = (bf16_t*)(ws + ub);
    bf16_t* Blo = Bhi + (size_t)H_DIM * D_IN;

    prep_w1_kernel<<<(H_DIM * D_IN) / 256, 256, 0, stream>>>(W1, Bhi, Blo);
    gemm_u_kernel<<<T / 16, 32, 0, stream>>>(X, Bhi, Blo, b1, U);
    scan_kernel<<<1, 32, 0, stream>>>(U, W1, W2, b2, out, T);
}
